// DprnnBlock_111669150284
// MI455X (gfx1250) — matrix operand order repaired and checked by reading, not compiled, not run on hardware
//
#include <hip/hip_runtime.h>
#include <hip/hip_bf16.h>
#include <math.h>

typedef _Float16 f16;
typedef __attribute__((ext_vector_type(16))) _Float16 v16h;
typedef __attribute__((ext_vector_type(8)))  _Float16 v8h;
typedef __attribute__((ext_vector_type(8)))  float    v8f;

#define BB 256
#define WW 128
#define CC 512
#define HD 256

__device__ __forceinline__ v8f wmma16x16x32(v16h a, v16h b, v8f c) {
    // D = A(16x32 f16) x B(32x16 f16) + C(16x16 f32)
    return __builtin_amdgcn_wmma_f32_16x16x32_f16(false, a, false, b, (short)0, c, false, false);
}

__device__ __forceinline__ v8f v8f_zero() {
    v8f z; for (int i = 0; i < 8; ++i) z[i] = 0.0f; return z;
}

__device__ __forceinline__ float hsig(float x) {
    return fminf(fmaxf(0.2f * x + 0.5f, 0.0f), 1.0f);
}

// ---------------------------------------------------------------------------
// f32 -> f16 conversion
// ---------------------------------------------------------------------------
__global__ void k_cvt_f16(const float* __restrict__ src, f16* __restrict__ dst, int n) {
    int i = blockIdx.x * blockDim.x + threadIdx.x;
    if (i < n) dst[i] = (f16)src[i];
}

// ---------------------------------------------------------------------------
// Generic f16 GEMM with bias: out[M,N] = A[M,K] @ B[K,N] + bias[N]
// Block: 256 threads (8 waves). Tile: 128(M) x 64(N), K step 32.
// Wave w owns M rows [w*16, w*16+16), 4 N-tiles of 16.
// ---------------------------------------------------------------------------
#define GBM 128
#define GBN 64
#define GBK 32

__global__ __launch_bounds__(256) void k_gemm_f16(
    const f16* __restrict__ A, int lda,
    const f16* __restrict__ B, int ldb,
    const float* __restrict__ bias,
    float* __restrict__ outF, f16* __restrict__ outH, int ldo,
    int K)
{
    __shared__ f16 As[GBM][GBK];   // 8 KB
    __shared__ f16 Bs[GBK][GBN];   // 4 KB

    const int tid   = threadIdx.x;
    const int wave  = tid >> 5;
    const int lane  = tid & 31;
    const int row16 = lane & 15;
    const int hi    = lane >> 4;
    const int m0    = blockIdx.x * GBM;
    const int n0    = blockIdx.y * GBN;

    v8f acc[4];
    for (int i = 0; i < 4; ++i) acc[i] = v8f_zero();

    const int rowA = tid >> 1, colA = (tid & 1) * 16;  // 128 rows x 32 cols
    const int rowB = tid >> 3, colB = (tid & 7) * 8;   // 32 rows x 64 cols

    for (int k0 = 0; k0 < K; k0 += GBK) {
        __syncthreads();
        *(v8h*)&As[rowA][colA]     = *(const v8h*)&A[(size_t)(m0 + rowA) * lda + k0 + colA];
        *(v8h*)&As[rowA][colA + 8] = *(const v8h*)&A[(size_t)(m0 + rowA) * lda + k0 + colA + 8];
        *(v8h*)&Bs[rowB][colB]     = *(const v8h*)&B[(size_t)(k0 + rowB) * ldb + n0 + colB];
        __syncthreads();

        v16h a;
        ((v8h*)&a)[0] = *(v8h*)&As[wave * 16 + row16][hi * 8];
        ((v8h*)&a)[1] = *(v8h*)&As[wave * 16 + row16][hi * 8 + 16];
        #pragma unroll
        for (int nt = 0; nt < 4; ++nt) {
            v16h b;
            for (int s = 0; s < 16; ++s)
                b[s] = Bs[s < 8 ? hi * 8 + s : 16 + hi * 8 + (s - 8)][nt * 16 + row16];
            acc[nt] = wmma16x16x32(a, b, acc[nt]);
        }
    }

    #pragma unroll
    for (int nt = 0; nt < 4; ++nt) {
        #pragma unroll
        for (int r = 0; r < 8; ++r) {
            int m = m0 + wave * 16 + r + hi * 8;
            int n = n0 + nt * 16 + row16;
            float v = acc[nt][r] + bias[n];
            if (outF) outF[(size_t)m * ldo + n] = v;
            else      outH[(size_t)m * ldo + n] = (f16)v;
        }
    }
}

// ---------------------------------------------------------------------------
// Intra bidirectional LSTM scan. grid = (16 batch tiles, 2 directions),
// block = 512 threads (16 waves). Each block owns 16 batch rows for all 128
// steps; per-row recurrence is independent -> no cross-block sync needed.
// Wave w owns hidden cols [w*16, w*16+16) and computes all 4 gate tiles.
// h (16x256) lives in LDS (f16); c lives in registers.
// ---------------------------------------------------------------------------
__global__ __launch_bounds__(512) void k_intra_scan(
    const f16* __restrict__ zxF, const f16* __restrict__ zxB,
    const f16* __restrict__ Uf,  const f16* __restrict__ Ub,
    f16* __restrict__ hs)
{
    __shared__ f16 hsh[16][HD];   // 8 KB

    const int dir  = blockIdx.y;
    const f16* zx  = dir ? zxB : zxF;
    const f16* U   = dir ? Ub  : Uf;
    const int n0   = blockIdx.x * 16;
    const int tid  = threadIdx.x;
    const int wave = tid >> 5;
    const int lane = tid & 31;
    const int row16 = lane & 15;
    const int hi    = lane >> 4;
    const int hid0  = wave * 16;

    // zero initial h: 4096 halves / 8 = 512 stores
    {
        v8h zh; for (int i = 0; i < 8; ++i) zh[i] = (f16)0;
        ((v8h*)&hsh[0][0])[tid] = zh;
    }
    float c[8];
    #pragma unroll
    for (int r = 0; r < 8; ++r) c[r] = 0.0f;
    __syncthreads();

    for (int s = 0; s < WW; ++s) {
        const int t = dir ? (WW - 1 - s) : s;

        // init accumulators with precomputed zx (input projection + bias)
        v8f acc[4];
        #pragma unroll
        for (int g = 0; g < 4; ++g) {
            acc[g] = v8f_zero();
            #pragma unroll
            for (int r = 0; r < 8; ++r) {
                int m   = r + hi * 8;
                int col = g * HD + hid0 + row16;
                acc[g][r] = (float)zx[((size_t)(n0 + m) * WW + t) * (4 * HD) + col];
            }
        }

        // zz += h_prev @ U   (K = 256, chunks of 32)
        for (int k0 = 0; k0 < HD; k0 += 32) {
            v16h a;
            ((v8h*)&a)[0] = *(v8h*)&hsh[row16][k0 + hi * 8];
            ((v8h*)&a)[1] = *(v8h*)&hsh[row16][k0 + hi * 8 + 16];
            #pragma unroll
            for (int g = 0; g < 4; ++g) {
                int nb = g * HD + hid0;
                v16h b;
                for (int s = 0; s < 16; ++s)
                    b[s] = U[(size_t)(k0 + (s < 8 ? hi * 8 + s : 16 + hi * 8 + (s - 8))) * (4 * HD) + nb + row16];
                acc[g] = wmma16x16x32(a, b, acc[g]);
            }
        }
        __syncthreads();   // all waves done reading h_prev

        #pragma unroll
        for (int r = 0; r < 8; ++r) {
            float iv = acc[0][r], fv = acc[1][r], gv = acc[2][r], ov = acc[3][r];
            float cn = hsig(fv) * c[r] + hsig(iv) * tanhf(gv);
            c[r] = cn;
            float hv = hsig(ov) * tanhf(cn);
            int m = r + hi * 8;
            f16 h16 = (f16)hv;
            hsh[m][hid0 + row16] = h16;
            // hs layout: [n, t, 512], fwd in cols 0..255, bwd in cols 256..511
            hs[((size_t)(n0 + m) * WW + t) * CC + dir * HD + hid0 + row16] = h16;
        }
        __syncthreads();   // h_new visible for next step
    }
}

// ---------------------------------------------------------------------------
// Fused inter LSTM (single step, L=1):
//   z = X@Wi + H0@Ui + bi ; gates -> c_new, h_new  (no z tensor in memory)
// grid = (32768/16, 512/128), block = 256 (8 waves). Wave owns one 16-wide
// hidden tile; its 4 accumulators are the i/f/g/o gate tiles.
// ---------------------------------------------------------------------------
__global__ __launch_bounds__(256) void k_inter_lstm(
    const f16* __restrict__ X,   // [32768,512] intra_out f16
    const f16* __restrict__ Wi,  // [512,2048]
    const f16* __restrict__ H0,  // [32768,512]
    const f16* __restrict__ Ui,  // [512,2048]
    const float* __restrict__ bi,   // [2048]
    const float* __restrict__ c0,   // [32768,512]
    float* __restrict__ hnew, float* __restrict__ cnew,
    f16* __restrict__ hnew16)
{
    const int tid   = threadIdx.x;
    const int wave  = tid >> 5;
    const int lane  = tid & 31;
    const int row16 = lane & 15;
    const int hi    = lane >> 4;
    const int m0    = blockIdx.x * 16;
    const int hb    = blockIdx.y * 128 + wave * 16;   // hidden-tile base

    v8f acc[4];
    for (int i = 0; i < 4; ++i) acc[i] = v8f_zero();

    #pragma unroll
    for (int ph = 0; ph < 2; ++ph) {
        const f16* A = ph ? H0 : X;
        const f16* B = ph ? Ui : Wi;
        for (int k0 = 0; k0 < CC; k0 += 32) {
            v16h a;
            ((v8h*)&a)[0] = *(const v8h*)&A[(size_t)(m0 + row16) * CC + k0 + hi * 8];
            ((v8h*)&a)[1] = *(const v8h*)&A[(size_t)(m0 + row16) * CC + k0 + hi * 8 + 16];
            #pragma unroll
            for (int g = 0; g < 4; ++g) {
                int nb = g * CC + hb;
                v16h b;
                for (int s = 0; s < 16; ++s)
                    b[s] = B[(size_t)(k0 + (s < 8 ? hi * 8 + s : 16 + hi * 8 + (s - 8))) * (4 * CC) + nb + row16];
                acc[g] = wmma16x16x32(a, b, acc[g]);
            }
        }
    }

    #pragma unroll
    for (int r = 0; r < 8; ++r) {
        int m = m0 + r + hi * 8;
        int n = hb + row16;
        float iv = acc[0][r] + bi[n];
        float fv = acc[1][r] + bi[CC + n];
        float gv = acc[2][r] + bi[2 * CC + n];
        float ov = acc[3][r] + bi[3 * CC + n];
        float c0v = c0[(size_t)m * CC + n];
        float cn = hsig(fv) * c0v + hsig(iv) * tanhf(gv);
        float hv = hsig(ov) * tanhf(cn);
        cnew[(size_t)m * CC + n]   = cn;
        hnew[(size_t)m * CC + n]   = hv;
        hnew16[(size_t)m * CC + n] = (f16)hv;
    }
}

// ---------------------------------------------------------------------------
// LayerNorm stats over (W,C) = 65536 elements per sample b. grid = 256.
// ---------------------------------------------------------------------------
__global__ __launch_bounds__(256) void k_ln_stats(const float* __restrict__ x,
                                                  float* __restrict__ stats)
{
    const int b = blockIdx.x;
    const float* p = x + (size_t)b * (WW * CC);
    float s = 0.0f, s2 = 0.0f;
    for (int i = threadIdx.x; i < WW * CC; i += 256) {
        float v = p[i]; s += v; s2 += v * v;
    }
    __shared__ float sh[256], sh2[256];
    sh[threadIdx.x] = s; sh2[threadIdx.x] = s2;
    __syncthreads();
    for (int off = 128; off > 0; off >>= 1) {
        if (threadIdx.x < off) {
            sh[threadIdx.x]  += sh[threadIdx.x + off];
            sh2[threadIdx.x] += sh2[threadIdx.x + off];
        }
        __syncthreads();
    }
    if (threadIdx.x == 0) {
        const float inv = 1.0f / (float)(WW * CC);
        float m = sh[0] * inv;
        float var = sh2[0] * inv - m * m;
        var = var < 0.0f ? 0.0f : var;
        stats[b * 2]     = m;
        stats[b * 2 + 1] = rsqrtf(var + 1e-3f);
    }
}

// intra_out = x + LN(pre)*g1 + b1 ; also emit f16 copy for the inter GEMM
__global__ void k_apply_intra(const float* __restrict__ xin, const float* __restrict__ pre,
                              const float* __restrict__ stats,
                              const float* __restrict__ g1, const float* __restrict__ b1,
                              float* __restrict__ out0, f16* __restrict__ ii16)
{
    size_t idx = (size_t)blockIdx.x * blockDim.x + threadIdx.x;
    int b  = (int)(idx >> 16);         // 65536 elems per sample
    int wc = (int)(idx & 65535);
    float m = stats[b * 2], r = stats[b * 2 + 1];
    float v = xin[idx] + (pre[idx] - m) * r * g1[wc] + b1[wc];
    out0[idx] = v;
    ii16[idx] = (f16)v;
}

// out += LN(pre)*g2 + b2
__global__ void k_apply_inter(const float* __restrict__ pre,
                              const float* __restrict__ stats,
                              const float* __restrict__ g2, const float* __restrict__ b2,
                              float* __restrict__ out0)
{
    size_t idx = (size_t)blockIdx.x * blockDim.x + threadIdx.x;
    int b  = (int)(idx >> 16);
    int wc = (int)(idx & 65535);
    float m = stats[b * 2], r = stats[b * 2 + 1];
    out0[idx] += (pre[idx] - m) * r * g2[wc] + b2[wc];
}

// ---------------------------------------------------------------------------
extern "C" void kernel_launch(void* const* d_in, const int* in_sizes, int n_in,
                              void* d_out, int out_size, void* d_ws, size_t ws_size,
                              hipStream_t stream)
{
    (void)in_sizes; (void)n_in; (void)out_size; (void)ws_size;

    const float* x    = (const float*)d_in[0];
    const float* h0f  = (const float*)d_in[1];   // (1, B*W, C)
    const float* c0f  = (const float*)d_in[2];
    const float* Wf   = (const float*)d_in[3];
    const float* Uf   = (const float*)d_in[4];
    const float* bf   = (const float*)d_in[5];
    const float* Wb   = (const float*)d_in[6];
    const float* Ub   = (const float*)d_in[7];
    const float* bb   = (const float*)d_in[8];
    const float* Wfc1 = (const float*)d_in[9];
    const float* bfc1 = (const float*)d_in[10];
    const float* g1   = (const float*)d_in[11];
    const float* b1   = (const float*)d_in[12];
    const float* Wi   = (const float*)d_in[13];
    const float* Ui   = (const float*)d_in[14];
    const float* bi   = (const float*)d_in[15];
    const float* Wfc2 = (const float*)d_in[16];
    const float* bfc2 = (const float*)d_in[17];
    const float* g2   = (const float*)d_in[18];
    const float* b2   = (const float*)d_in[19];

    const size_t NTOT = (size_t)BB * WW * CC;          // 16,777,216
    float* outMain = (float*)d_out;                    // [B,1,W,C]
    float* outH    = outMain + NTOT;                   // h_new [B*W,C]
    float* outC    = outH + NTOT;                      // c_new [B*W,C]

    // ---- carve workspace ----
    char* w = (char*)d_ws;
    auto alloc = [&](size_t bytes) -> void* {
        void* p = (void*)w; w += (bytes + 255) & ~(size_t)255; return p;
    };
    f16*   x16    = (f16*)  alloc(NTOT * 2);
    f16*   h016   = (f16*)  alloc(NTOT * 2);
    f16*   Wf16   = (f16*)  alloc((size_t)CC * 4 * HD * 2);
    f16*   Wb16   = (f16*)  alloc((size_t)CC * 4 * HD * 2);
    f16*   Uf16   = (f16*)  alloc((size_t)HD * 4 * HD * 2);
    f16*   Ub16   = (f16*)  alloc((size_t)HD * 4 * HD * 2);
    f16*   Wfc116 = (f16*)  alloc((size_t)CC * CC * 2);
    f16*   Wi16   = (f16*)  alloc((size_t)CC * 4 * CC * 2);
    f16*   Ui16   = (f16*)  alloc((size_t)CC * 4 * CC * 2);
    f16*   Wfc216 = (f16*)  alloc((size_t)CC * CC * 2);
    f16*   zxF    = (f16*)  alloc((size_t)BB * WW * 4 * HD * 2);
    f16*   zxB    = (f16*)  alloc((size_t)BB * WW * 4 * HD * 2);
    f16*   hs16   = (f16*)  alloc(NTOT * 2);
    float* pre    = (float*)alloc(NTOT * 4);           // fc1 out, reused for fc2 out
    f16*   ii16   = (f16*)  alloc(NTOT * 2);           // intra_out f16
    f16*   hn16   = (f16*)  alloc(NTOT * 2);           // h_new f16
    float* stats  = (float*)alloc(1024 * 4);
    float* stats2 = stats + 512;

    auto cvt = [&](const float* s, f16* d, int n) {
        k_cvt_f16<<<(n + 255) / 256, 256, 0, stream>>>(s, d, n);
    };

    // ---- 1. f16 conversions ----
    cvt(x,    x16,    (int)NTOT);
    cvt(h0f,  h016,   (int)NTOT);
    cvt(Wf,   Wf16,   CC * 4 * HD);
    cvt(Wb,   Wb16,   CC * 4 * HD);
    cvt(Uf,   Uf16,   HD * 4 * HD);
    cvt(Ub,   Ub16,   HD * 4 * HD);
    cvt(Wfc1, Wfc116, CC * CC);
    cvt(Wi,   Wi16,   CC * 4 * CC);
    cvt(Ui,   Ui16,   CC * 4 * CC);
    cvt(Wfc2, Wfc216, CC * CC);

    const int Mrows = BB * WW;   // 32768

    // ---- 2. intra input projections: zx = x @ W + b  (M=32768, N=1024, K=512)
    {
        dim3 g(Mrows / GBM, (4 * HD) / GBN);
        k_gemm_f16<<<g, 256, 0, stream>>>(x16, CC, Wf16, 4 * HD, bf,
                                          nullptr, zxF, 4 * HD, CC);
        k_gemm_f16<<<g, 256, 0, stream>>>(x16, CC, Wb16, 4 * HD, bb,
                                          nullptr, zxB, 4 * HD, CC);
    }

    // ---- 3. bidirectional recurrent scan over W=128 ----
    {
        dim3 g(BB / 16, 2);
        k_intra_scan<<<g, 512, 0, stream>>>(zxF, zxB, Uf16, Ub16, hs16);
    }

    // ---- 4. fc1: pre = hs @ Wfc1 + bfc1  (M=32768, N=512, K=512)
    {
        dim3 g(Mrows / GBM, CC / GBN);
        k_gemm_f16<<<g, 256, 0, stream>>>(hs16, CC, Wfc116, CC, bfc1,
                                          pre, nullptr, CC, CC);
    }

    // ---- 5. LN over (W,C) + residual -> intra_out (to d_out) + f16 copy ----
    k_ln_stats<<<BB, 256, 0, stream>>>(pre, stats);
    k_apply_intra<<<(unsigned)(NTOT / 256), 256, 0, stream>>>(
        x, pre, stats, g1, b1, outMain, ii16);

    // ---- 6. fused inter LSTM step -> h_new, c_new ----
    {
        dim3 g(Mrows / 16, CC / 128);
        k_inter_lstm<<<g, 256, 0, stream>>>(ii16, Wi16, h016, Ui16, bi, c0f,
                                            outH, outC, hn16);
    }

    // ---- 7. fc2: pre = h_new @ Wfc2 + bfc2 ----
    {
        dim3 g(Mrows / GBM, CC / GBN);
        k_gemm_f16<<<g, 256, 0, stream>>>(hn16, CC, Wfc216, CC, bfc2,
                                          pre, nullptr, CC, CC);
    }

    // ---- 8. LN + final residual add into d_out ----
    k_ln_stats<<<BB, 256, 0, stream>>>(pre, stats2);
    k_apply_inter<<<(unsigned)(NTOT / 256), 256, 0, stream>>>(
        pre, stats2, g2, b2, outMain);
}
